// PINN_7842610282672
// MI455X (gfx1250) — matrix operand order repaired and checked by reading, not compiled, not run on hardware
//
#include <hip/hip_runtime.h>
#include <math.h>
#include <stdint.h>

// ---------------------------------------------------------------------------
// PINN DeepSets fused kernel for gfx1250 (MI455X, wave32, WMMA + TDM)
//   phi: 3 -> 100 -> 100 -> 5   (per particle, mean over 32)
//   psi: 1 -> 100 -> 100 -> 5   (per pair,    sum over 496)
//   rho: 10 -> 100 -> 100 -> 1
// Hidden GEMMs run on v_wmma_f32_16x16x32_f16 (f16 in, f32 accumulate).
// One workgroup (8 wave32) per batch element; everything stays in LDS.
// Weight staging uses the Tensor Data Mover (tensor_load_to_lds, 6-arg form).
// ---------------------------------------------------------------------------

#define NPART   32
#define NPAIR   496          // 32*31/2, == 31 * 16 exactly
#define HID     100
#define DL      5
#define LDK     128          // K padded to 4 WMMA k-steps of 32
#define NPADT   7            // 7 N-tiles of 16 -> 112 padded columns
#define WAVES   8

typedef __attribute__((ext_vector_type(16))) _Float16 v16h;
typedef __attribute__((ext_vector_type(8)))  _Float16 v8h;
typedef __attribute__((ext_vector_type(8)))  float    v8f;
typedef __attribute__((ext_vector_type(4)))  unsigned int v4u;
typedef __attribute__((ext_vector_type(8)))  int          v8i;
typedef __attribute__((ext_vector_type(4)))  int          v4i;

union AF { v16h v; v8h h[2]; };

// Branch-free GELU (exact erf via Abramowitz-Stegun 7.1.26, |err|<1.5e-7).
// No exec-mask divergence: ~12 FMA + v_rcp + v_exp per call.
__device__ __forceinline__ float gelu_exact(float x) {
  float ax = fabsf(x) * 0.70710678118654752440f;     // |x|/sqrt(2)
  float t  = __builtin_amdgcn_rcpf(__builtin_fmaf(0.3275911f, ax, 1.0f));
  float p  = __builtin_fmaf(1.061405429f, t, -1.453152027f);
  p = __builtin_fmaf(p, t, 1.421413741f);
  p = __builtin_fmaf(p, t, -0.284496736f);
  p = __builtin_fmaf(p, t, 0.254829592f);
  p = p * t;
  float e  = __expf(-ax * ax);
  float erf_ax = __builtin_fmaf(-p, e, 1.0f);        // erf(|x|/sqrt2)
  float erfv = copysignf(erf_ax, x);
  return 0.5f * x * (1.0f + erfv);
}

// A fragment (16x32 f16, M x K) from row-major f16 [rows][row_stride] buffer.
// Lanes 0-15: row = mr+l, K chunks {kb+0..7, kb+16..23}
// Lanes 16-31: row = mr+(l-16), K chunks {kb+8..15, kb+24..31}
__device__ __forceinline__ v16h load_a(const _Float16* base, int row_stride,
                                       int mr, int kb, int l) {
  AF a;
  const _Float16* p = base + (mr + (l & 15)) * row_stride + kb + ((l >> 4) << 3);
  a.h[0] = *(const v8h*)(p);
  a.h[1] = *(const v8h*)(p + 16);
  return a.v;
}

// B fragment (32x16 f16, K x N) from weight buffer stored n-major [n][LDK]
// (B[k][n] = W[n][k], i.e. the row-major torch weight, zero padded).
// Lane n = nb + (l&15); lanes 0-15 hold K {kb+0..7, kb+16..23}, lanes 16-31 hold {kb+8..15, kb+24..31}.
__device__ __forceinline__ v16h load_b(const _Float16* base, int nb, int kb, int l) {
  AF b;
  const _Float16* p = base + (nb + (l & 15)) * LDK + kb + ((l >> 4) << 3);
  b.h[0] = *(const v8h*)(p);
  b.h[1] = *(const v8h*)(p + 16);
  return b.v;
}

// Fused hidden tail: h1 = gelu(h0 @ W1^T + b1); partial = h1 @ W2^T  (bias of
// layer 2 folded in later analytically). Column sums of the DL outputs are
// accumulated into sumOut[0..4] with LDS float atomics (ds_add_f32).
__device__ __forceinline__ void block_mlp_tail(
    const _Float16* sA, int mtiles,
    const _Float16* sW1, const _Float16* sW2,
    const float* __restrict__ b1,
    _Float16* h1scr_all, float* sumOut) {
  const int tid  = threadIdx.x;
  const int wave = tid >> 5;
  const int l    = tid & 31;
  _Float16* h1 = h1scr_all + wave * 16 * LDK;   // per-wave 16x128 f16 scratch

  for (int mt = wave; mt < mtiles; mt += WAVES) {
    const int mr = mt * 16;
    // A fragments for all four k-steps of this M-tile (reused across N-tiles)
    v16h a0 = load_a(sA, LDK, mr,  0, l);
    v16h a1 = load_a(sA, LDK, mr, 32, l);
    v16h a2 = load_a(sA, LDK, mr, 64, l);
    v16h a3 = load_a(sA, LDK, mr, 96, l);

    // ---- layer 1: 7 N-tiles x 4 k-steps of WMMA ----
    for (int nt = 0; nt < NPADT; ++nt) {
      v8f acc = {};
      acc = __builtin_amdgcn_wmma_f32_16x16x32_f16(false, a0, false,
              load_b(sW1, nt * 16,  0, l), (short)0, acc, false, false);
      acc = __builtin_amdgcn_wmma_f32_16x16x32_f16(false, a1, false,
              load_b(sW1, nt * 16, 32, l), (short)0, acc, false, false);
      acc = __builtin_amdgcn_wmma_f32_16x16x32_f16(false, a2, false,
              load_b(sW1, nt * 16, 64, l), (short)0, acc, false, false);
      acc = __builtin_amdgcn_wmma_f32_16x16x32_f16(false, a3, false,
              load_b(sW1, nt * 16, 96, l), (short)0, acc, false, false);
      // C/D layout: lane holds col n = nt*16 + (l&15), rows v + (l>>4)*8
      const int coln  = nt * 16 + (l & 15);
      const float bia = (coln < HID) ? b1[coln] : 0.0f;
      const int rbase = (l >> 4) << 3;
#pragma unroll
      for (int v = 0; v < 8; ++v) {
        float h = (coln < HID) ? gelu_exact(acc[v] + bia) : 0.0f;
        h1[(rbase + v) * LDK + coln] = (_Float16)h;
      }
    }
    // zero the K-pad columns 112..127 of the scratch tile
    {
      v8h z = {};
      *(v8h*)(h1 + (l & 15) * LDK + 112 + ((l >> 4) << 3)) = z;
    }

    // ---- layer 2: 1 N-tile x 4 k-steps ----
    v8f acc2 = {};
    acc2 = __builtin_amdgcn_wmma_f32_16x16x32_f16(false, load_a(h1, LDK, 0,  0, l),
             false, load_b(sW2, 0,  0, l), (short)0, acc2, false, false);
    acc2 = __builtin_amdgcn_wmma_f32_16x16x32_f16(false, load_a(h1, LDK, 0, 32, l),
             false, load_b(sW2, 0, 32, l), (short)0, acc2, false, false);
    acc2 = __builtin_amdgcn_wmma_f32_16x16x32_f16(false, load_a(h1, LDK, 0, 64, l),
             false, load_b(sW2, 0, 64, l), (short)0, acc2, false, false);
    acc2 = __builtin_amdgcn_wmma_f32_16x16x32_f16(false, load_a(h1, LDK, 0, 96, l),
             false, load_b(sW2, 0, 96, l), (short)0, acc2, false, false);

    const int n = l & 15;
    if (n < DL) {
      float s = acc2[0] + acc2[1] + acc2[2] + acc2[3]
              + acc2[4] + acc2[5] + acc2[6] + acc2[7];
      atomicAdd(&sumOut[n], s);   // ds_add_f32
    }
  }
}

// ---------------------------------------------------------------------------
// Weight prep: pack W1 (100x100) and W2 (5x100) of psi and phi into
// zero-padded f16, n-major [n][LDK] layouts in workspace:
//   [ B1psi 112x128 | B2psi 16x128 | B1phi 112x128 | B2phi 16x128 ]
// ---------------------------------------------------------------------------
#define SZ_B1 (112 * 128)
#define SZ_B2 (16 * 128)
#define WBYTES (2 * (SZ_B1 + SZ_B2) * 2)   // 65536 bytes of packed f16

__global__ void pinn_prep_weights(const float* __restrict__ psiW1,
                                  const float* __restrict__ psiW2,
                                  const float* __restrict__ phiW1,
                                  const float* __restrict__ phiW2,
                                  _Float16* __restrict__ wsh) {
  const int total = 2 * (SZ_B1 + SZ_B2);
  for (int idx = blockIdx.x * blockDim.x + threadIdx.x; idx < total;
       idx += gridDim.x * blockDim.x) {
    int t = idx;
    float v = 0.0f;
    if (t < SZ_B1) {
      int n = t >> 7, k = t & 127;
      if (n < HID && k < HID) v = psiW1[n * HID + k];
    } else if (t < SZ_B1 + SZ_B2) {
      int u = t - SZ_B1, n = u >> 7, k = u & 127;
      if (n < DL && k < HID) v = psiW2[n * HID + k];
    } else if (t < 2 * SZ_B1 + SZ_B2) {
      int u = t - (SZ_B1 + SZ_B2), n = u >> 7, k = u & 127;
      if (n < HID && k < HID) v = phiW1[n * HID + k];
    } else {
      int u = t - (2 * SZ_B1 + SZ_B2), n = u >> 7, k = u & 127;
      if (n < DL && k < HID) v = phiW2[n * HID + k];
    }
    wsh[idx] = (_Float16)v;
  }
}

// ---------------------------------------------------------------------------
// Main fused kernel: one workgroup per batch element.
// ---------------------------------------------------------------------------
__global__ __launch_bounds__(256) void pinn_fused(
    const float* __restrict__ x,
    const float* __restrict__ phiW0, const float* __restrict__ phiB0,
    const float* __restrict__ phiB1, const float* __restrict__ phiB2,
    const float* __restrict__ psiW0, const float* __restrict__ psiB0,
    const float* __restrict__ psiB1, const float* __restrict__ psiB2,
    const float* __restrict__ rhoW0, const float* __restrict__ rhoB0,
    const float* __restrict__ rhoW1, const float* __restrict__ rhoB1,
    const float* __restrict__ rhoW2, const float* __restrict__ rhoB2,
    const _Float16* __restrict__ wsh, float* __restrict__ out) {
  extern __shared__ char smem[];
  const int tid = threadIdx.x;
  const int b   = blockIdx.x;

  // ---- dynamic LDS carve-up (total ~236.8 KB of the 320 KB WGP LDS) ----
  _Float16* sW1psi = (_Float16*)smem;                // 112x128
  _Float16* sW2psi = sW1psi + SZ_B1;                 // 16x128
  _Float16* sW1phi = sW2psi + SZ_B2;                 // 112x128
  _Float16* sW2phi = sW1phi + SZ_B1;                 // 16x128
  _Float16* h0psi  = sW2phi + SZ_B2;                 // 496x128 f16
  _Float16* h1scr  = h0psi + NPAIR * LDK;            // 8 waves x 16x128 f16
  _Float16* h0phi  = h1scr + WAVES * 16 * LDK;       // 32x128 f16
  float*    rbuf   = (float*)(h0phi + NPART * LDK);  // 496 pair distances
  float*    xs     = rbuf + NPAIR;                   // 96 = 32x3 coords
  float*    psiSum = xs + 96;                        // 8
  float*    phiSum = psiSum + 8;                     // 8
  float*    r0     = phiSum + 8;                     // 112
  float*    r1     = r0 + 112;                       // 112
  float*    redAcc = r1 + 112;                       // 1

  const float* xb = x + (size_t)b * (NPART * 3);
  __builtin_prefetch(xb, 0, 3);                      // global_prefetch_b8

  // ---- stage packed f16 weights (64 KB) into LDS via the Tensor Data Mover
#if __has_builtin(__builtin_amdgcn_tensor_load_to_lds)
  if (tid < 32) {     // single wave issues the TDM descriptor (EXEC ignored)
    const uint64_t ga = (uint64_t)(uintptr_t)wsh;
    const uint32_t lds_off = (uint32_t)(uintptr_t)sW1psi;
    // D# group 0: count=1, lds_addr, global_addr[56:0], type=2 ("image")
    v4u g0;
    g0[0] = 1u;                                       // count=1, user mode
    g0[1] = lds_off;                                  // LDS byte address
    g0[2] = (uint32_t)(ga & 0xFFFFFFFFu);             // global addr [95:64]
    g0[3] = (uint32_t)((ga >> 32) & 0x01FFFFFFu) | (2u << 30);  // [120:96]+type
    // D# group 1: data_size=8B, 1-D tile of 8192 elements (= 64 KB)
    v8i g1;
    g1[0] = (3 << 16);          // workgroup_mask=0, data_size=3 (8 bytes)
    g1[1] = (int)(8192u << 16); // tensor_dim0[15:0] in bits 63:48
    g1[2] = (int)(1u << 16);    // tensor_dim0[31:16]=0 | tensor_dim1[15:0]=1
    g1[3] = (int)(8192u << 16); // tensor_dim1[31:16]=0 | tile_dim0=8192
    g1[4] = 1;                  // tile_dim1=1, tile_dim2=0
    g1[5] = 8192;               // tensor_dim0_stride[31:0]
    g1[6] = (int)(8192u << 16); // stride0[47:32]=0 | tensor_dim1_stride[15:0]
    g1[7] = 0;                  // tensor_dim1_stride[47:16]
    v4i gz4 = {0, 0, 0, 0};            // groups 2/3 unused (<=2-D tensor)
    v8i gz8 = {0, 0, 0, 0, 0, 0, 0, 0};
    __builtin_amdgcn_tensor_load_to_lds(g0, g1, gz4, gz4, gz8, 0);
    __builtin_amdgcn_s_wait_tensorcnt(0);
  }
#else
  {
    const uint4* src = (const uint4*)wsh;
    uint4*       dst = (uint4*)sW1psi;
    for (int i = tid; i < WBYTES / 16; i += 256) dst[i] = src[i];
  }
#endif
  if (tid < NPART * 3) xs[tid] = xb[tid];
  if (tid < 8) { psiSum[tid] = 0.0f; phiSum[tid] = 0.0f; }
  if (tid == 0) redAcc[0] = 0.0f;
  __syncthreads();

  // ---- pair distances r_ij for the 496 upper-triangle pairs ----
  for (int p = tid; p < NPAIR; p += 256) {
    int i = 0, rem = p, cnt = NPART - 1;
    while (rem >= cnt) { rem -= cnt; --cnt; ++i; }
    int j = i + 1 + rem;
    float dx = xs[i * 3 + 0] - xs[j * 3 + 0];
    float dy = xs[i * 3 + 1] - xs[j * 3 + 1];
    float dz = xs[i * 3 + 2] - xs[j * 3 + 2];
    rbuf[p] = sqrtf(dx * dx + dy * dy + dz * dz);
  }
  __syncthreads();

  // ---- input layers (VALU): psi h0 = gelu(r*w0 + b0), phi h0 = gelu(x.w0+b0)
  for (int t = tid; t < NPAIR * LDK; t += 256) {
    int p = t >> 7, c = t & 127;
    float v = 0.0f;
    if (c < HID) v = gelu_exact(rbuf[p] * psiW0[c] + psiB0[c]);
    h0psi[t] = (_Float16)v;
  }
  for (int t = tid; t < NPART * LDK; t += 256) {
    int i = t >> 7, c = t & 127;
    float v = 0.0f;
    if (c < HID) {
      float a = xs[i * 3 + 0] * phiW0[c * 3 + 0]
              + xs[i * 3 + 1] * phiW0[c * 3 + 1]
              + xs[i * 3 + 2] * phiW0[c * 3 + 2] + phiB0[c];
      v = gelu_exact(a);
    }
    h0phi[t] = (_Float16)v;
  }
  __syncthreads();

  // ---- WMMA hidden layers + per-batch reductions ----
  block_mlp_tail(h0psi, NPAIR / 16, sW1psi, sW2psi, psiB1, h1scr, psiSum);
  block_mlp_tail(h0phi, NPART / 16, sW1phi, sW2phi, phiB1, h1scr, phiSum);
  __syncthreads();

  // ---- rho head: 10 -> 100 -> 100 -> 1 (one row, VALU) ----
  // layer-2 biases of phi/psi folded in here:
  //   phi_mean[q] = phiSum[q]/32 + phiB2[q];  psi_tot[q] = psiSum[q] + 496*psiB2[q]
  if (tid < 112) {
    float v = 0.0f;
    if (tid < HID) {
      float in[10];
#pragma unroll
      for (int q = 0; q < DL; ++q) in[q]      = phiSum[q] * (1.0f / NPART) + phiB2[q];
#pragma unroll
      for (int q = 0; q < DL; ++q) in[DL + q] = psiSum[q] + (float)NPAIR * psiB2[q];
      float a = rhoB0[tid];
#pragma unroll
      for (int q = 0; q < 10; ++q) a += in[q] * rhoW0[tid * 10 + q];
      v = gelu_exact(a);
    }
    r0[tid] = v;
  }
  __syncthreads();
  if (tid < 112) {
    float v = 0.0f;
    if (tid < HID) {
      float a = rhoB1[tid];
      for (int q = 0; q < HID; ++q) a += r0[q] * rhoW1[tid * HID + q];
      v = gelu_exact(a);
    }
    r1[tid] = v;
  }
  __syncthreads();
  if (tid < HID) atomicAdd(redAcc, r1[tid] * rhoW2[tid]);
  __syncthreads();
  if (tid == 0) out[b] = redAcc[0] + rhoB2[0];
}

// ---------------------------------------------------------------------------
extern "C" void kernel_launch(void* const* d_in, const int* in_sizes, int n_in,
                              void* d_out, int out_size, void* d_ws, size_t ws_size,
                              hipStream_t stream) {
  (void)in_sizes; (void)n_in; (void)out_size; (void)ws_size;
  const float* x     = (const float*)d_in[0];
  const float* phiW0 = (const float*)d_in[1];
  const float* phiB0 = (const float*)d_in[2];
  const float* phiW1 = (const float*)d_in[3];
  const float* phiB1 = (const float*)d_in[4];
  const float* phiW2 = (const float*)d_in[5];
  const float* phiB2 = (const float*)d_in[6];
  const float* psiW0 = (const float*)d_in[7];
  const float* psiB0 = (const float*)d_in[8];
  const float* psiW1 = (const float*)d_in[9];
  const float* psiB1 = (const float*)d_in[10];
  const float* psiW2 = (const float*)d_in[11];
  const float* psiB2 = (const float*)d_in[12];
  const float* rhoW0 = (const float*)d_in[13];
  const float* rhoB0 = (const float*)d_in[14];
  const float* rhoW1 = (const float*)d_in[15];
  const float* rhoB1 = (const float*)d_in[16];
  const float* rhoW2 = (const float*)d_in[17];
  const float* rhoB2 = (const float*)d_in[18];
  _Float16* wsh = (_Float16*)d_ws;

  pinn_prep_weights<<<64, 256, 0, stream>>>(psiW1, psiW2, phiW1, phiW2, wsh);

  // dynamic LDS: f16 regions (32768+63488+16384+4096 halves) + 833 floats
  size_t shbytes = (size_t)(2 * (SZ_B1 + SZ_B2) + NPAIR * LDK +
                            WAVES * 16 * LDK + NPART * LDK) * sizeof(_Float16)
                 + (size_t)(NPAIR + 96 + 8 + 8 + 112 + 112 + 1) * sizeof(float);
  shbytes = (shbytes + 63) & ~(size_t)63;

  pinn_fused<<<4096, 256, shbytes, stream>>>(
      x, phiW0, phiB0, phiB1, phiB2,
      psiW0, psiB0, psiB1, psiB2,
      rhoW0, rhoB0, rhoW1, rhoB1, rhoW2, rhoB2,
      wsh, (float*)d_out);
}
